// GNNRecommender_56642028700253
// MI455X (gfx1250) — compile-verified, matrix operand order checked by reading, not run on hardware
//
#include <hip/hip_runtime.h>
#include <hip/hip_bf16.h>

#define NUM_USERS 20000
#define NUM_ITEMS 30000
#define N_NODES   (NUM_USERS + NUM_ITEMS)
#define N_EDGES   1000000
#define FEAT      19
#define EMB       64
#define HID       128
#define NHEADS    8
#define BSZ       256

typedef __attribute__((ext_vector_type(16))) _Float16 v16h;
typedef __attribute__((ext_vector_type(8)))  float    v8f;

// Fragment index helpers for 16-bit 16x32 WMMA operands:
//   k-in-chunk (0..31) -> half = (k>>3)&1 ; idx16 = ((k>>4)<<3) | (k&7)
//   inverse: idx16 j, half -> kin = ((j>>3)<<4) | (j&7) | (half<<3)
__device__ __forceinline__ int frag_half(int kin) { return (kin >> 3) & 1; }
__device__ __forceinline__ int frag_idx (int kin) { return ((kin >> 4) << 3) | (kin & 7); }

// ---------------------------------------------------------------------------
// 1. Feature projection: h0[n, j] = x[n,:19] . feat_W[j,:19] + feat_b[j]
// ---------------------------------------------------------------------------
__global__ void feat_proj_kernel(const float* __restrict__ x,
                                 const float* __restrict__ W,
                                 const float* __restrict__ b,
                                 float* __restrict__ h)
{
    int tid = blockIdx.x * blockDim.x + threadIdx.x;
    if (tid >= N_NODES * EMB) return;
    int n = tid / EMB;
    int j = tid % EMB;
    float s = b[j];
    const float* xp = x + (size_t)n * FEAT;
    const float* wp = W + (size_t)j * FEAT;
#pragma unroll
    for (int f = 0; f < FEAT; ++f) s += xp[f] * wp[f];
    h[tid] = s;
}

// ---------------------------------------------------------------------------
// 2. Degree (in-degree over col + self loop) and dinv = rsqrt(deg)
// ---------------------------------------------------------------------------
__global__ void init_deg_kernel(float* __restrict__ deg)
{
    int i = blockIdx.x * blockDim.x + threadIdx.x;
    if (i < N_NODES) deg[i] = 1.0f;  // self loop
}

__global__ void edge_deg_kernel(const int* __restrict__ ei, float* __restrict__ deg)
{
    int e = blockIdx.x * blockDim.x + threadIdx.x;
    if (e >= N_EDGES) return;
    int c = ei[N_EDGES + e];
    atomicAdd(&deg[c], 1.0f);
}

__global__ void dinv_kernel(const float* __restrict__ deg, float* __restrict__ dinv)
{
    int i = blockIdx.x * blockDim.x + threadIdx.x;
    if (i < N_NODES) dinv[i] = __frsqrt_rn(deg[i]);
}

// ---------------------------------------------------------------------------
// 3a. agg init with self-loop term: agg[i,f] = X[i,f] * dinv[i]^2
// ---------------------------------------------------------------------------
__global__ void init_agg_kernel(const float* __restrict__ X,
                                const float* __restrict__ dinv,
                                float* __restrict__ agg, int Kd)
{
    long long tid = (long long)blockIdx.x * blockDim.x + threadIdx.x;
    long long total = (long long)N_NODES * Kd;
    if (tid >= total) return;
    int n = (int)(tid / Kd);
    float d = dinv[n];
    agg[tid] = X[tid] * d * d;
}

// ---------------------------------------------------------------------------
// 3b. edge scatter: agg[col] += X[row] * (dinv[row]*dinv[col]*ew)
// ---------------------------------------------------------------------------
__global__ void scatter_edges_kernel(const float* __restrict__ X,
                                     const int* __restrict__ ei,
                                     const float* __restrict__ ew,
                                     const float* __restrict__ dinv,
                                     float* __restrict__ agg, int Kd)
{
    int chunks = Kd >> 2;
    long long tid = (long long)blockIdx.x * blockDim.x + threadIdx.x;
    long long total = (long long)N_EDGES * chunks;
    if (tid >= total) return;
    int e  = (int)(tid / chunks);
    int f0 = (int)(tid % chunks) * 4;
    int r = ei[e];
    int c = ei[N_EDGES + e];
    float coeff = dinv[r] * dinv[c] * ew[e];
    const float* xp = X + (size_t)r * Kd + f0;
    float* ap = agg + (size_t)c * Kd + f0;
    atomicAdd(ap + 0, xp[0] * coeff);
    atomicAdd(ap + 1, xp[1] * coeff);
    atomicAdd(ap + 2, xp[2] * coeff);
    atomicAdd(ap + 3, xp[3] * coeff);
}

// ---------------------------------------------------------------------------
// 3c. pack activations into WMMA A-fragment order (f16), optional /deg.
//     One thread per (tile, chunk, lane): 8 aligned float2 gathers from the
//     row-major source, one aligned 32B vector store of the fragment block.
//     dst layout: [(tile*CH + chunk)][lane(0..31)][idx16(0..15)]
// ---------------------------------------------------------------------------
__global__ void pack_act_kernel(const float* __restrict__ src,
                                const float* __restrict__ deg,
                                _Float16* __restrict__ dst,
                                int Kd, int use_deg)
{
    int CH = Kd >> 5;
    int tid = blockIdx.x * blockDim.x + threadIdx.x;
    int total = (N_NODES / 16) * CH * 32;
    if (tid >= total) return;
    int lane = tid & 31;
    int tc   = tid >> 5;          // tile*CH + chunk
    int c    = tc % CH;
    int tile = tc / CH;
    int half = lane >> 4;
    int mr   = lane & 15;
    int n    = tile * 16 + mr;
    float scale = use_deg ? (1.0f / deg[n]) : 1.0f;
    const float* sp = src + (size_t)n * Kd + c * 32;
    v16h outv;
#pragma unroll
    for (int j = 0; j < 16; j += 2) {
        int kin = (((j >> 3) << 4) | (j & 7)) + half * 8;   // even; pair contiguous
        float2 f = *(const float2*)(sp + kin);
        outv[j]     = (_Float16)(f.x * scale);
        outv[j + 1] = (_Float16)(f.y * scale);
    }
    *(v16h*)(dst + (size_t)tc * 512 + lane * 16) = outv;
}

// ---------------------------------------------------------------------------
// 3d. pack weight W[128 x K] (row-major f32) into WMMA B-fragment order (f16).
//     dst layout: [((chunk*128 + n)*2 + half)][idx16(0..15)]
// ---------------------------------------------------------------------------
__global__ void pack_w_kernel(const float* __restrict__ W,
                              _Float16* __restrict__ dst, int K)
{
    int tid = blockIdx.x * blockDim.x + threadIdx.x;
    if (tid >= HID * K) return;
    int n = tid / K;
    int k = tid % K;
    int c = k >> 5, kin = k & 31;
    size_t d = (((size_t)c * HID + n) * 2 + frag_half(kin)) * 16 + frag_idx(kin);
    dst[d] = (_Float16)W[tid];
}

// ---------------------------------------------------------------------------
// 4. WMMA GEMM: OUT[M x 128] = act( A[M x K](f16 packed) * Wpack + bias )
//    One wave per 16-row tile, 8 waves/block. Packed weights staged in LDS.
//    B fragments software-pipelined one tile ahead so ds_load latency
//    overlaps the matrix pipe.
// ---------------------------------------------------------------------------
template <int KDIM>
__global__ __launch_bounds__(256) void gemm_wmma_kernel(
    const _Float16* __restrict__ Apack, const _Float16* __restrict__ Wpack,
    const float* __restrict__ bias, float* __restrict__ OUT,
    int M, int relu)
{
    constexpr int CH = KDIM / 32;
    __shared__ _Float16 smem[KDIM * HID];   // packed B fragments

    // cooperative stage of packed weights into LDS (16B vector copies)
    {
        uint4* s4 = (uint4*)smem;
        const uint4* g4 = (const uint4*)Wpack;
        const int n16 = (KDIM * HID * 2) / 16;
        for (int i = threadIdx.x; i < n16; i += blockDim.x) s4[i] = g4[i];
    }
    __syncthreads();

    int wave = blockIdx.x * (blockDim.x >> 5) + (threadIdx.x >> 5);
    int lane = threadIdx.x & 31;
    int row0 = wave * 16;
    if (row0 < M) {                       // wave-uniform: EXEC all-1s inside
        int half = lane >> 4;
        int mr   = lane & 15;

        v8f acc[8];
#pragma unroll
        for (int t = 0; t < 8; ++t) acc[t] = (v8f){0,0,0,0,0,0,0,0};

        const _Float16* abase = Apack + ((size_t)wave * CH) * 512 + lane * 16;

        auto loadB = [&](int c, int t) -> v16h {
            int n = t * 16 + mr;
            return *(const v16h*)(smem + (((c * HID + n) * 2 + half) << 4));
        };

        v16h bcur = loadB(0, 0);
#pragma unroll
        for (int c = 0; c < CH; ++c) {
            v16h a = *(const v16h*)(abase + (size_t)c * 512);
#pragma unroll
            for (int t = 0; t < 8; ++t) {
                v16h bnext = bcur;
                if (t < 7)            bnext = loadB(c, t + 1);
                else if (c + 1 < CH)  bnext = loadB(c + 1, 0);
                acc[t] = __builtin_amdgcn_wmma_f32_16x16x32_f16(
                    false, a, false, bcur, (short)0, acc[t], false, false);
                bcur = bnext;
            }
        }

        // D layout: lane%16 = N col, vgpr v -> M = v + (lane/16)*8
#pragma unroll
        for (int t = 0; t < 8; ++t) {
            int n = t * 16 + mr;
            float bv = bias[n];
#pragma unroll
            for (int v = 0; v < 8; ++v) {
                int mm = row0 + v + half * 8;
                if (mm < M) {
                    float val = acc[t][v] + bv;
                    if (relu) val = fmaxf(val, 0.0f);
                    OUT[(size_t)mm * HID + n] = val;
                }
            }
        }
    }
}

// ---------------------------------------------------------------------------
// 6. Query projection with gather: q[b,n] = h[base+idx[b],:] . Wq[n,:] + bq[n]
// ---------------------------------------------------------------------------
__global__ void qproj_kernel(const float* __restrict__ h,
                             const int* __restrict__ idx, int base,
                             const float* __restrict__ Wq,
                             const float* __restrict__ bq,
                             float* __restrict__ q)
{
    int tid = blockIdx.x * blockDim.x + threadIdx.x;
    if (tid >= BSZ * HID) return;
    int b = tid >> 7;
    int n = tid & 127;
    int row = base + idx[b];
    float s = bq[n];
    const float* hp = h + (size_t)row * HID;
    const float* wp = Wq + (size_t)n * HID;
    for (int k = 0; k < HID; ++k) s += hp[k] * wp[k];
    q[tid] = s;
}

// ---------------------------------------------------------------------------
// 7. Attention: one wave per (query b, head h); two-pass softmax; wave32
//    reductions via __shfl_xor.
// ---------------------------------------------------------------------------
__global__ __launch_bounds__(256) void attend_kernel(
    const float* __restrict__ q, const float* __restrict__ Kmat,
    const float* __restrict__ Vmat, int Mkeys, float* __restrict__ out)
{
    int wave = threadIdx.x >> 5;
    int lane = threadIdx.x & 31;
    int pair = blockIdx.x * 8 + wave;   // 0 .. 2047
    int b = pair >> 3;
    int h = pair & 7;

    const float* qp = q + (size_t)b * HID + h * 16;
    float qv[16];
#pragma unroll
    for (int d = 0; d < 16; ++d) qv[d] = qp[d];
    const float scale = 0.25f;          // 1/sqrt(16)

    float smax = -1e30f;
    for (int m = lane; m < Mkeys; m += 32) {
        const float* kp = Kmat + (size_t)m * HID + h * 16;
        float s = 0.0f;
#pragma unroll
        for (int d = 0; d < 16; ++d) s += qv[d] * kp[d];
        smax = fmaxf(smax, s * scale);
    }
#pragma unroll
    for (int off = 16; off >= 1; off >>= 1)
        smax = fmaxf(smax, __shfl_xor(smax, off, 32));

    float sumexp = 0.0f;
    float acc[16];
#pragma unroll
    for (int d = 0; d < 16; ++d) acc[d] = 0.0f;
    for (int m = lane; m < Mkeys; m += 32) {
        const float* kp = Kmat + (size_t)m * HID + h * 16;
        const float* vp = Vmat + (size_t)m * HID + h * 16;
        float s = 0.0f;
#pragma unroll
        for (int d = 0; d < 16; ++d) s += qv[d] * kp[d];
        float e = __expf(s * scale - smax);
        sumexp += e;
#pragma unroll
        for (int d = 0; d < 16; ++d) acc[d] += e * vp[d];
    }
#pragma unroll
    for (int off = 16; off >= 1; off >>= 1) {
        sumexp += __shfl_xor(sumexp, off, 32);
#pragma unroll
        for (int d = 0; d < 16; ++d) acc[d] += __shfl_xor(acc[d], off, 32);
    }
    if (lane == 0) {
        float inv = 1.0f / sumexp;
        float* op = out + (size_t)b * HID + h * 16;
        for (int d = 0; d < 16; ++d) op[d] = acc[d] * inv;
    }
}

// ---------------------------------------------------------------------------
// 8. Small dense layer: out[b, n] = act( in[b,:K] . W[n,:K] + bias[n] )
// ---------------------------------------------------------------------------
__global__ void linear_kernel(const float* __restrict__ in,
                              const float* __restrict__ W,
                              const float* __restrict__ bias,
                              float* __restrict__ out,
                              int M, int N, int K, int ldo, int relu)
{
    int tid = blockIdx.x * blockDim.x + threadIdx.x;
    if (tid >= M * N) return;
    int b = tid / N;
    int n = tid % N;
    float s = bias[n];
    const float* ip = in + (size_t)b * K;
    const float* wp = W + (size_t)n * K;
    for (int k = 0; k < K; ++k) s += ip[k] * wp[k];
    if (relu) s = fmaxf(s, 0.0f);
    out[(size_t)b * ldo + n] = s;
}

// ---------------------------------------------------------------------------
// 9. Head: p = sigmoid(z2 . pW3 + pb3); r = (p*rW + rb)*4 + 1
// ---------------------------------------------------------------------------
__global__ void head_kernel(const float* __restrict__ z2,
                            const float* __restrict__ pW3,
                            const float* __restrict__ pb3,
                            const float* __restrict__ rW,
                            const float* __restrict__ rb,
                            float* __restrict__ out)
{
    int b = blockIdx.x * blockDim.x + threadIdx.x;
    if (b >= BSZ) return;
    float s = pb3[0];
    for (int k = 0; k < 32; ++k) s += z2[b * 32 + k] * pW3[k];
    float p = 1.0f / (1.0f + __expf(-s));
    out[b] = (p * rW[0] + rb[0]) * 4.0f + 1.0f;
}

// ---------------------------------------------------------------------------
// Launch
// ---------------------------------------------------------------------------
extern "C" void kernel_launch(void* const* d_in, const int* in_sizes, int n_in,
                              void* d_out, int out_size, void* d_ws, size_t ws_size,
                              hipStream_t stream)
{
    const float* x          = (const float*)d_in[0];
    const int*   edge_index = (const int*)  d_in[1];
    const float* edge_attr  = (const float*)d_in[2];
    const int*   user_idx   = (const int*)  d_in[3];
    const int*   item_idx   = (const int*)  d_in[4];
    const float* feat_W     = (const float*)d_in[5];
    const float* feat_b     = (const float*)d_in[6];
    const float* gW0        = (const float*)d_in[7];
    const float* gb0        = (const float*)d_in[8];
    const float* gW1        = (const float*)d_in[9];
    const float* gb1        = (const float*)d_in[10];
    const float* gW2        = (const float*)d_in[11];
    const float* gb2        = (const float*)d_in[12];
    const float* ipw        = (const float*)d_in[13];
    const float* ipb        = (const float*)d_in[14];
    const float* out_w      = (const float*)d_in[15];
    const float* out_b      = (const float*)d_in[16];
    const float* pW1        = (const float*)d_in[17];
    const float* pb1        = (const float*)d_in[18];
    const float* pW2        = (const float*)d_in[19];
    const float* pb2        = (const float*)d_in[20];
    const float* pW3        = (const float*)d_in[21];
    const float* pb3        = (const float*)d_in[22];
    const float* rW         = (const float*)d_in[23];
    const float* rb         = (const float*)d_in[24];
    float* out = (float*)d_out;
    (void)in_sizes; (void)n_in; (void)out_size; (void)ws_size;

    // ---- workspace carving ----
    char* ws = (char*)d_ws;
    size_t off = 0;
    auto carve = [&](size_t bytes) -> char* {
        char* p = ws + off;
        off += (bytes + 255) & ~(size_t)255;
        return p;
    };
    float*    bufA   = (float*)   carve((size_t)N_NODES * HID * 4);
    float*    bufB   = (float*)   carve((size_t)N_NODES * HID * 4);
    float*    agg    = (float*)   carve((size_t)N_NODES * HID * 4);
    _Float16* apack  = (_Float16*)carve((size_t)N_NODES * HID * 2);
    float*    deg    = (float*)   carve((size_t)N_NODES * 4);
    float*    dinv   = (float*)   carve((size_t)N_NODES * 4);
    float*    Kall   = (float*)   carve((size_t)N_NODES * HID * 4);
    float*    Vall   = (float*)   carve((size_t)N_NODES * HID * 4);
    _Float16* wp0    = (_Float16*)carve((size_t)HID * EMB * 2);
    _Float16* wp1    = (_Float16*)carve((size_t)HID * HID * 2);
    _Float16* wp2    = (_Float16*)carve((size_t)HID * HID * 2);
    _Float16* wpk    = (_Float16*)carve((size_t)HID * HID * 2);
    _Float16* wpv    = (_Float16*)carve((size_t)HID * HID * 2);
    float*    q_u    = (float*)   carve((size_t)BSZ * HID * 4);
    float*    q_i    = (float*)   carve((size_t)BSZ * HID * 4);
    float*    attn_u = (float*)   carve((size_t)BSZ * HID * 4);
    float*    attn_i = (float*)   carve((size_t)BSZ * HID * 4);
    float*    zcat   = (float*)   carve((size_t)BSZ * 2 * HID * 4);
    float*    z1     = (float*)   carve((size_t)BSZ * HID * 4);
    float*    z2     = (float*)   carve((size_t)BSZ * 32 * 4);

    const int T = 256;
    auto grid1 = [](long long total, int block) -> unsigned int {
        return (unsigned int)((total + block - 1) / block);
    };

    // 0. pack all WMMA weights to f16 fragment order
    pack_w_kernel<<<grid1(HID * EMB, T), T, 0, stream>>>(gW0, wp0, EMB);
    pack_w_kernel<<<grid1(HID * HID, T), T, 0, stream>>>(gW1, wp1, HID);
    pack_w_kernel<<<grid1(HID * HID, T), T, 0, stream>>>(gW2, wp2, HID);
    pack_w_kernel<<<grid1(HID * HID, T), T, 0, stream>>>(ipw + HID * HID, wpk, HID);
    pack_w_kernel<<<grid1(HID * HID, T), T, 0, stream>>>(ipw + 2 * HID * HID, wpv, HID);

    // 1. feature projection -> bufA (50000 x 64)
    feat_proj_kernel<<<grid1((long long)N_NODES * EMB, T), T, 0, stream>>>(
        x, feat_W, feat_b, bufA);

    // 2. degree + dinv
    init_deg_kernel<<<grid1(N_NODES, T), T, 0, stream>>>(deg);
    edge_deg_kernel<<<grid1(N_EDGES, T), T, 0, stream>>>(edge_index, deg);
    dinv_kernel<<<grid1(N_NODES, T), T, 0, stream>>>(deg, dinv);

    // 3. three GCN layers
    int tiles = (N_NODES + 15) / 16;
    unsigned int ggrid = (unsigned int)((tiles + 7) / 8);
    auto run_layer = [&](const float* Xin, int Kd, const _Float16* Wp,
                         const float* bias, float* Yout) {
        long long tot   = (long long)N_NODES * Kd;
        long long etot  = (long long)N_EDGES * (Kd / 4);
        long long ptot  = (long long)(N_NODES / 16) * (Kd / 32) * 32;
        init_agg_kernel<<<grid1(tot, T), T, 0, stream>>>(Xin, dinv, agg, Kd);
        scatter_edges_kernel<<<grid1(etot, T), T, 0, stream>>>(
            Xin, edge_index, edge_attr, dinv, agg, Kd);
        pack_act_kernel<<<grid1(ptot, T), T, 0, stream>>>(agg, deg, apack, Kd, 1);
        if (Kd == 64)
            gemm_wmma_kernel<64><<<ggrid, T, 0, stream>>>(apack, Wp, bias, Yout,
                                                          N_NODES, 1);
        else
            gemm_wmma_kernel<128><<<ggrid, T, 0, stream>>>(apack, Wp, bias, Yout,
                                                           N_NODES, 1);
    };
    run_layer(bufA,  64, wp0, gb0, bufB);
    run_layer(bufB, 128, wp1, gb1, bufA);
    run_layer(bufA, 128, wp2, gb2, bufB);   // h3 = bufB

    // 4. K/V projections over all nodes (users+items share Wk/Wv)
    long long ptot128 = (long long)(N_NODES / 16) * (HID / 32) * 32;
    pack_act_kernel<<<grid1(ptot128, T), T, 0, stream>>>(bufB, deg, apack, HID, 0);
    gemm_wmma_kernel<128><<<ggrid, T, 0, stream>>>(apack, wpk, ipb + HID,
                                                   Kall, N_NODES, 0);
    gemm_wmma_kernel<128><<<ggrid, T, 0, stream>>>(apack, wpv, ipb + 2 * HID,
                                                   Vall, N_NODES, 0);

    // 5. query projections (gather + Wq)
    qproj_kernel<<<grid1((long long)BSZ * HID, T), T, 0, stream>>>(
        bufB, user_idx, 0, ipw, ipb, q_u);
    qproj_kernel<<<grid1((long long)BSZ * HID, T), T, 0, stream>>>(
        bufB, item_idx, NUM_USERS, ipw, ipb, q_i);

    // 6. attention (wave per (b, head); 2048 pairs -> 256 blocks of 8 waves)
    attend_kernel<<<256, T, 0, stream>>>(q_u, Kall, Vall, NUM_USERS, attn_u);
    attend_kernel<<<256, T, 0, stream>>>(q_i, Kall + (size_t)NUM_USERS * HID,
                                         Vall + (size_t)NUM_USERS * HID,
                                         NUM_ITEMS, attn_i);

    // 7. output projections write straight into concat buffer zcat[256 x 256]
    linear_kernel<<<grid1((long long)BSZ * HID, T), T, 0, stream>>>(
        attn_u, out_w, out_b, zcat,        BSZ, HID, HID, 2 * HID, 0);
    linear_kernel<<<grid1((long long)BSZ * HID, T), T, 0, stream>>>(
        attn_i, out_w, out_b, zcat + HID,  BSZ, HID, HID, 2 * HID, 0);

    // 8. MLP
    linear_kernel<<<grid1((long long)BSZ * HID, T), T, 0, stream>>>(
        zcat, pW1, pb1, z1, BSZ, HID, 2 * HID, HID, 1);
    linear_kernel<<<grid1((long long)BSZ * 32, T), T, 0, stream>>>(
        z1, pW2, pb2, z2, BSZ, 32, HID, 32, 1);

    // 9. head -> d_out (256 f32)
    head_kernel<<<1, T, 0, stream>>>(z2, pW3, pb3, rW, rb, out);
}
